// LAUDecoder_51427938402965
// MI455X (gfx1250) — matrix operand order repaired and checked by reading, not compiled, not run on hardware
//
#include <hip/hip_runtime.h>

// ---------------------------------------------------------------------------
// LAU decoder step for MI455X (gfx1250, wave32, WMMA).
//
// Roofline: ~143 GFLOP total, dominated by enc_context @ Ws2a^T (137 GFLOP).
// Mandatory HBM traffic ~0.75 GB -> ~32us at 23.3 TB/s; bf16 WMMA compute is
// far faster than that, so everything here is bandwidth-bound. We therefore:
//   * fuse the [B,S,A] attn_h tensor away (saves 2x268MB of HBM round trip),
//   * convert f32 -> bf16 only in registers/LDS (packed v_cvt, RTNE) and
//     accumulate in f32 with v_wmma_f32_16x16x32_bf16,
//   * keep all intermediates small ([64,1024] / [64,3072]) in d_ws.
// ---------------------------------------------------------------------------

typedef __attribute__((ext_vector_type(16))) __bf16        v16bf;
typedef __attribute__((ext_vector_type(4)))  __bf16        v4bf;
typedef __attribute__((ext_vector_type(8)))  float         v8f;
typedef __attribute__((ext_vector_type(4)))  float         v4f;
typedef __attribute__((ext_vector_type(4)))  unsigned int  v4u;

#define BB   64
#define SS   1024
#define HH   1024
#define H3   3072
#define NLAY 4

__device__ __forceinline__ float sigmoidf_(float x) {
  return 1.0f / (1.0f + __expf(-x));
}

// Pack 8 floats -> 8 bf16 (16 bytes) using native packed converts.
union Pack8 { v4bf h[2]; v4u q; };
__device__ __forceinline__ v4u cvt8(v4f a, v4f b) {
  Pack8 p;
  p.h[0] = __builtin_convertvector(a, v4bf);
  p.h[1] = __builtin_convertvector(b, v4bf);
  return p.q;
}

// Fragment union: 16 bf16 = 32 bytes = what one lane owns of a 16x32 (A) or
// 32x16 (B) bf16 WMMA operand.
union Frag {
  v4u   q[2];
  v16bf v;
};

// ---------------------------------------------------------------------------
// Generic skinny GEMM:  Y[64,N] = X[64,K] @ W[N,K]^T + bias[N]
// N multiple of 128, K multiple of 32.  8 waves/WG; wave w owns a 16-col
// stripe; each wave computes 4 M-tiles (64 rows).  A/B staged in LDS as bf16.
//
// bf16 operand layouts used here (A and B share contiguous k order):
//   A (16x32): lane l -> row (l&15); elems j     = K 16*(l>>4)+j
//   B (32x16): lane l -> col (l&15); elems j     = K 16*(l>>4)+j
//   C/D      : vgpr v, lane l -> m = v + 8*(l>>4), n = (l&15)
// ---------------------------------------------------------------------------
__global__ void __launch_bounds__(256)
gemm_bias_kernel(const float* __restrict__ X, const float* __restrict__ W,
                 const float* __restrict__ bias, float* __restrict__ Y,
                 int N, int K) {
  __shared__ __align__(16) __bf16 As[64][32];
  __shared__ __align__(16) __bf16 Bs[128][32];

  const int tid   = threadIdx.x;
  const int lane  = tid & 31;
  const int wave  = tid >> 5;        // 0..7
  const int half  = lane >> 4;       // 0/1
  const int l16   = lane & 15;
  const int nBase = blockIdx.x * 128;

  v8f acc[4] = {};

  for (int k0 = 0; k0 < K; k0 += 32) {
    // Stage A tile 64x32 (2048 elems, 8/thread, one b128 LDS store each).
    {
      int idx = tid * 8;
      int r = idx >> 5, c = idx & 31;
      const float* src = X + (size_t)r * K + k0 + c;
      if (k0 + 32 < K) __builtin_prefetch(src + 32, 0, 1);
      v4f a0 = *(const v4f*)(src);
      v4f a1 = *(const v4f*)(src + 4);
      *(v4u*)(&As[r][c]) = cvt8(a0, a1);
    }
    // Stage W tile 128x32 (4096 elems, 16/thread, two b128 LDS stores).
    {
      int idx = tid * 16;
      int r = idx >> 5, c = idx & 31;
      const float* src = W + (size_t)(nBase + r) * K + k0 + c;
      if (k0 + 32 < K) __builtin_prefetch(src + 32, 0, 1);
      v4f w0 = *(const v4f*)(src);
      v4f w1 = *(const v4f*)(src + 4);
      v4f w2 = *(const v4f*)(src + 8);
      v4f w3 = *(const v4f*)(src + 12);
      *(v4u*)(&Bs[r][c])     = cvt8(w0, w1);
      *(v4u*)(&Bs[r][c + 8]) = cvt8(w2, w3);
    }
    __syncthreads();

    // B fragment for this wave's 16-col stripe.
    Frag fb;
    fb.q[0] = *(const v4u*)(&Bs[wave * 16 + l16][16 * half]);
    fb.q[1] = *(const v4u*)(&Bs[wave * 16 + l16][16 * half + 8]);

    #pragma unroll
    for (int mt = 0; mt < 4; mt++) {
      Frag fa;
      int row = mt * 16 + l16;
      fa.q[0] = *(const v4u*)(&As[row][16 * half]);
      fa.q[1] = *(const v4u*)(&As[row][16 * half + 8]);
      acc[mt] = __builtin_amdgcn_wmma_f32_16x16x32_bf16(
          false, fa.v, false, fb.v, (short)0, acc[mt], false, false);
    }
    __syncthreads();
  }

  const int n = nBase + wave * 16 + l16;
  const float bv = bias ? bias[n] : 0.0f;
  #pragma unroll
  for (int mt = 0; mt < 4; mt++) {
    #pragma unroll
    for (int v = 0; v < 8; v++) {
      int m = mt * 16 + v + 8 * half;
      Y[(size_t)m * N + n] = acc[mt][v] + bv;
    }
  }
}

// ---------------------------------------------------------------------------
// Fused attention logits: for 16 rows (b, s0..s0+15) compute
//   row = enc[b,s,:] @ Ws2a^T + bs2a                  (WMMA, K=C=1024)
//   v   = LN(row)*att_sg+att_sb + hp[b,:]             (hp pre-LN'd h0 proj)
//   logit[b,s] = tanh(v) . Wa2o + ba2o   (masked)
// 16x1024 accumulator block parked in 64KB dynamic LDS for the row-wise LN.
// 8 waves; wave w owns cols [w*128, w*128+128) as 8 WMMA N-tiles.
// ---------------------------------------------------------------------------
__global__ void __launch_bounds__(256)
attn_logit_kernel(const float* __restrict__ enc,    // [B,S,C]
                  const float* __restrict__ Ws2a,   // [A,C]
                  const float* __restrict__ bs2a,   // [A]
                  const float* __restrict__ sg, const float* __restrict__ sb,
                  const float* __restrict__ hp,     // [B,A]
                  const float* __restrict__ Wa2o,   // [A]
                  const float* __restrict__ ba2o,   // [1]
                  const unsigned char* __restrict__ mask, // [B,S]
                  float* __restrict__ logit)        // [B,S]
{
  extern __shared__ float outBuf[];                 // [16][1024] = 64KB

  const int tid  = threadIdx.x;
  const int lane = tid & 31;
  const int wave = tid >> 5;
  const int half = lane >> 4;
  const int l16  = lane & 15;
  const int b    = blockIdx.x >> 6;                 // S/16 = 64 s-tiles per b
  const int s0   = (blockIdx.x & 63) * 16;

  v8f acc[8] = {};

  const float* arow = enc + ((size_t)(b * SS + s0 + l16)) * HH;

  for (int k0 = 0; k0 < HH; k0 += 32) {
    // A fragment straight from global (enc rows are K-contiguous).
    Frag fa;
    {
      const float* ap = arow + k0 + 16 * half;
      v4f a0 = *(const v4f*)(ap);
      v4f a1 = *(const v4f*)(ap + 4);
      v4f a2 = *(const v4f*)(ap + 8);
      v4f a3 = *(const v4f*)(ap + 12);
      fa.q[0] = cvt8(a0, a1);
      fa.q[1] = cvt8(a2, a3);
    }
    #pragma unroll
    for (int nt = 0; nt < 8; nt++) {
      const int n = wave * 128 + nt * 16 + l16;
      const float* wp = Ws2a + (size_t)n * HH + k0 + 16 * half;
      Frag fb;
      v4f w0 = *(const v4f*)(wp);
      v4f w1 = *(const v4f*)(wp + 4);
      v4f w2 = *(const v4f*)(wp + 8);
      v4f w3 = *(const v4f*)(wp + 12);
      fb.q[0] = cvt8(w0, w1);
      fb.q[1] = cvt8(w2, w3);
      acc[nt] = __builtin_amdgcn_wmma_f32_16x16x32_bf16(
          false, fa.v, false, fb.v, (short)0, acc[nt], false, false);
    }
  }

  // Park accumulators (+ bias) in LDS for the row-wise LN/tanh/dot epilogue.
  #pragma unroll
  for (int nt = 0; nt < 8; nt++) {
    const int col = wave * 128 + nt * 16 + l16;
    const float bv = bs2a[col];
    #pragma unroll
    for (int v = 0; v < 8; v++) {
      const int m = v + 8 * half;
      outBuf[m * 1024 + col] = acc[nt][v] + bv;
    }
  }
  __syncthreads();

  const float invA = 1.0f / 1024.0f;
  const float* hpr = hp + (size_t)b * HH;
  for (int rr = 0; rr < 2; rr++) {
    const int r = wave * 2 + rr;                    // 8 waves x 2 rows = 16
    const float* row = outBuf + r * 1024;
    float sum = 0.f, sq = 0.f;
    for (int c = lane; c < 1024; c += 32) { float x = row[c]; sum += x; sq += x * x; }
    #pragma unroll
    for (int off = 16; off; off >>= 1) { sum += __shfl_down(sum, off); sq += __shfl_down(sq, off); }
    sum = __shfl(sum, 0); sq = __shfl(sq, 0);
    const float mean = sum * invA;
    const float rstd = rsqrtf(sq * invA - mean * mean + 1e-5f);
    float dot = 0.f;
    for (int c = lane; c < 1024; c += 32) {
      float v = (row[c] - mean) * rstd * sg[c] + sb[c] + hpr[c];
      dot += tanhf(v) * Wa2o[c];
    }
    #pragma unroll
    for (int off = 16; off; off >>= 1) dot += __shfl_down(dot, off);
    if (lane == 0) {
      const int s = s0 + r;
      const float lg = dot + ba2o[0];
      logit[(size_t)b * SS + s] = mask[(size_t)b * SS + s] ? lg : -__builtin_huge_valf();
    }
  }
}

// ---------------------------------------------------------------------------
// Row-wise LayerNorm with gain/bias; one WG per row; in-place safe.
// ---------------------------------------------------------------------------
__global__ void __launch_bounds__(256)
rowln_kernel(float* __restrict__ y, const float* __restrict__ x,
             const float* __restrict__ g, const float* __restrict__ b, int N) {
  const int row = blockIdx.x, tid = threadIdx.x;
  const int lane = tid & 31, wave = tid >> 5;
  const float* xr = x + (size_t)row * N;
  float* yr = y + (size_t)row * N;
  __shared__ float redS[8], redQ[8];
  float sum = 0.f, sq = 0.f;
  for (int c = tid; c < N; c += 256) { float v = xr[c]; sum += v; sq += v * v; }
  #pragma unroll
  for (int off = 16; off; off >>= 1) { sum += __shfl_down(sum, off); sq += __shfl_down(sq, off); }
  if (lane == 0) { redS[wave] = sum; redQ[wave] = sq; }
  __syncthreads();
  if (tid == 0) {
    float s = 0.f, q = 0.f;
    for (int i = 0; i < 8; i++) { s += redS[i]; q += redQ[i]; }
    redS[0] = s; redQ[0] = q;
  }
  __syncthreads();
  const float invN = 1.0f / (float)N;
  const float mean = redS[0] * invN;
  const float rstd = rsqrtf(redQ[0] * invN - mean * mean + 1e-5f);
  for (int c = tid; c < N; c += 256) yr[c] = (xr[c] - mean) * rstd * g[c] + b[c];
}

// ---------------------------------------------------------------------------
// LAU gating pointwise combine (all operands already LayerNorm'd).
// ---------------------------------------------------------------------------
__global__ void __launch_bounds__(256)
cell_gate_kernel(const float* __restrict__ gx, const float* __restrict__ gh,  // [B,3H]
                 const float* __restrict__ px, const float* __restrict__ pxh,
                 const float* __restrict__ phh, const float* __restrict__ h,  // [B,H]
                 float* __restrict__ out) {
  const int i = blockIdx.x * blockDim.x + threadIdx.x;   // B*H threads
  const int b = i >> 10, j = i & 1023;
  const float* gxr = gx + (size_t)b * H3;
  const float* ghr = gh + (size_t)b * H3;
  const float r = sigmoidf_(gxr[j]        + ghr[j]);
  const float z = sigmoidf_(gxr[j + 1024] + ghr[j + 1024]);
  const float g = sigmoidf_(gxr[j + 2048] + ghr[j + 2048]);
  const float Hx = px[i], xh = pxh[i], hh = phh[i], hv = h[i];
  const float hm = tanhf((1.0f - r) * xh + r * hh);
  out[i] = ((1.0f - z) * hv + z * hm) * (1.0f - g) + g * Hx;
}

// ---------------------------------------------------------------------------
// Softmax over S per batch row.
// ---------------------------------------------------------------------------
__global__ void __launch_bounds__(256)
softmax_kernel(float* __restrict__ p, const float* __restrict__ logit, int S) {
  const int b = blockIdx.x, tid = threadIdx.x;
  const int lane = tid & 31, wave = tid >> 5;
  const float* lr = logit + (size_t)b * S;
  float* pr = p + (size_t)b * S;
  __shared__ float red[8];
  float mx = -3.402823466e38f;
  for (int s = tid; s < S; s += 256) mx = fmaxf(mx, lr[s]);
  #pragma unroll
  for (int off = 16; off; off >>= 1) mx = fmaxf(mx, __shfl_down(mx, off));
  if (lane == 0) red[wave] = mx;
  __syncthreads();
  if (tid == 0) { float m = red[0]; for (int i = 1; i < 8; i++) m = fmaxf(m, red[i]); red[0] = m; }
  __syncthreads();
  mx = red[0];
  __syncthreads();
  float sum = 0.f;
  for (int s = tid; s < S; s += 256) { float e = __expf(lr[s] - mx); pr[s] = e; sum += e; }
  #pragma unroll
  for (int off = 16; off; off >>= 1) sum += __shfl_down(sum, off);
  if (lane == 0) red[wave] = sum;
  __syncthreads();
  if (tid == 0) { float s = 0.f; for (int i = 0; i < 8; i++) s += red[i]; red[0] = s; }
  __syncthreads();
  const float inv = 1.0f / red[0];
  for (int s = tid; s < S; s += 256) pr[s] *= inv;
}

// ---------------------------------------------------------------------------
// attn[b,c] = sum_s p[b,s] * enc[b,s,c]  (coalesced over c across lanes)
// ---------------------------------------------------------------------------
__global__ void __launch_bounds__(256)
attnctx_kernel(float* __restrict__ out, const float* __restrict__ p,
               const float* __restrict__ enc, int S, int C) {
  const int b = blockIdx.x >> 2;                    // 4 c-tiles of 256
  const int c = (blockIdx.x & 3) * 256 + threadIdx.x;
  const float* pb = p + (size_t)b * S;
  const float* eb = enc + (size_t)b * S * C + c;
  float acc = 0.f;
  for (int s = 0; s < S; s++) acc += pb[s] * eb[(size_t)s * C];
  out[(size_t)b * C + c] = acc;
}

// proj = tanh(a + b + c), all operands pre-LN'd.
__global__ void __launch_bounds__(256)
readout_kernel(float* __restrict__ out, const float* __restrict__ a,
               const float* __restrict__ b, const float* __restrict__ c) {
  const int i = blockIdx.x * blockDim.x + threadIdx.x;
  out[i] = tanhf(a[i] + b[i] + c[i]);
}

// ---------------------------------------------------------------------------
// Host orchestration
// ---------------------------------------------------------------------------
extern "C" void kernel_launch(void* const* d_in, const int* in_sizes, int n_in,
                              void* d_out, int out_size, void* d_ws, size_t ws_size,
                              hipStream_t stream) {
  (void)in_sizes; (void)n_in; (void)out_size; (void)ws_size;

  const float* emb    = (const float*)d_in[0];
  const float* hm     = (const float*)d_in[1];
  const float* enc    = (const float*)d_in[2];
  const float* Wig    = (const float*)d_in[3];
  const float* big    = (const float*)d_in[4];
  const float* Whg    = (const float*)d_in[5];
  const float* bhg    = (const float*)d_in[6];
  const float* Wii    = (const float*)d_in[7];
  const float* bii    = (const float*)d_in[8];
  const float* Wih    = (const float*)d_in[9];
  const float* bih    = (const float*)d_in[10];
  const float* Whh    = (const float*)d_in[11];
  const float* bhh    = (const float*)d_in[12];
  const float* g3     = (const float*)d_in[13];
  const float* b3     = (const float*)d_in[14];
  const float* g1     = (const float*)d_in[15];
  const float* b1     = (const float*)d_in[16];
  const float* Wh2a   = (const float*)d_in[17];
  const float* bh2a   = (const float*)d_in[18];
  const float* Ws2a   = (const float*)d_in[19];
  const float* bs2a   = (const float*)d_in[20];
  const float* Wa2o   = (const float*)d_in[21];
  const float* ba2o   = (const float*)d_in[22];
  const float* att_hg = (const float*)d_in[23];
  const float* att_hb = (const float*)d_in[24];
  const float* att_sg = (const float*)d_in[25];
  const float* att_sb = (const float*)d_in[26];
  const float* We2o   = (const float*)d_in[27];
  const float* be2o   = (const float*)d_in[28];
  const float* Wh2o   = (const float*)d_in[29];
  const float* bh2o   = (const float*)d_in[30];
  const float* Wc2o   = (const float*)d_in[31];
  const float* bc2o   = (const float*)d_in[32];
  const float* eg     = (const float*)d_in[33];
  const float* eb     = (const float*)d_in[34];
  const float* hg     = (const float*)d_in[35];
  const float* hb     = (const float*)d_in[36];
  const float* cg     = (const float*)d_in[37];
  const float* cb     = (const float*)d_in[38];
  const unsigned char* mask = (const unsigned char*)d_in[39];

  float* outp = (float*)d_out;                // proj [64,1024]
  float* hid  = outp + (size_t)BB * HH;       // stacked hid [4,64,1024]

  float* ws = (float*)d_ws;
  auto take = [&](size_t n) { float* p = ws; ws += n; return p; };
  float* w_h0  = take((size_t)BB * HH);
  float* w_gx  = take((size_t)BB * H3);
  float* w_gh  = take((size_t)BB * H3);
  float* w_px  = take((size_t)BB * HH);
  float* w_pxh = take((size_t)BB * HH);
  float* w_phh = take((size_t)BB * HH);
  float* w_hp  = take((size_t)BB * HH);
  float* w_lg  = take((size_t)BB * SS);
  float* w_pr  = take((size_t)BB * SS);
  float* w_at  = take((size_t)BB * HH);
  float* w_r1  = take((size_t)BB * HH);
  float* w_r2  = take((size_t)BB * HH);
  float* w_r3  = take((size_t)BB * HH);

  auto gemm = [&](const float* X, const float* W, const float* bv, float* Y, int N, int K) {
    gemm_bias_kernel<<<dim3(N / 128), dim3(256), 0, stream>>>(X, W, bv, Y, N, K);
  };
  auto ln = [&](float* y, const float* x, const float* g, const float* bv, int rows, int N) {
    rowln_kernel<<<dim3(rows), dim3(256), 0, stream>>>(y, x, g, bv, N);
  };

  auto cellf = [&](int i, const float* x, const float* h, float* out) {
    const size_t s3 = (size_t)H3 * HH;     // 3H x H weight stride
    const size_t s1 = (size_t)HH * HH;     // H x H weight stride
    const size_t o3 = (size_t)i * 2 * H3;  // g3/b3 stride
    const size_t o1 = (size_t)i * 3 * HH;  // g1/b1 stride
    gemm(x, Wig + (size_t)i * s3, big + (size_t)i * H3, w_gx, H3, HH);
    ln(w_gx, w_gx, g3 + o3,        b3 + o3,        BB, H3);
    gemm(h, Whg + (size_t)i * s3, bhg + (size_t)i * H3, w_gh, H3, HH);
    ln(w_gh, w_gh, g3 + o3 + H3,   b3 + o3 + H3,   BB, H3);
    gemm(x, Wii + (size_t)i * s1, bii + (size_t)i * HH, w_px, HH, HH);
    ln(w_px, w_px, g1 + o1,        b1 + o1,        BB, HH);
    gemm(x, Wih + (size_t)i * s1, bih + (size_t)i * HH, w_pxh, HH, HH);
    ln(w_pxh, w_pxh, g1 + o1 + HH, b1 + o1 + HH,   BB, HH);
    gemm(h, Whh + (size_t)i * s1, bhh + (size_t)i * HH, w_phh, HH, HH);
    ln(w_phh, w_phh, g1 + o1 + 2 * HH, b1 + o1 + 2 * HH, BB, HH);
    cell_gate_kernel<<<dim3((BB * HH) / 256), dim3(256), 0, stream>>>(
        w_gx, w_gh, w_px, w_pxh, w_phh, h, out);
  };

  // 1) h0 = cell(0, emb, hidden_m[0])
  cellf(0, emb, hm, w_h0);

  // 2) hp = LN(h0 @ Wh2a^T + bh2a) * att_hg + att_hb
  gemm(w_h0, Wh2a, bh2a, w_hp, HH, HH);
  ln(w_hp, w_hp, att_hg, att_hb, BB, HH);

  // 3) fused attn_h -> logits (64KB dynamic LDS accumulator block)
  attn_logit_kernel<<<dim3(BB * (SS / 16)), dim3(256), 16 * 1024 * sizeof(float), stream>>>(
      enc, Ws2a, bs2a, att_sg, att_sb, w_hp, Wa2o, ba2o, mask, w_lg);

  // 4) softmax over S
  softmax_kernel<<<dim3(BB), dim3(256), 0, stream>>>(w_pr, w_lg, SS);

  // 5) attn context = p @ enc
  attnctx_kernel<<<dim3(BB * 4), dim3(256), 0, stream>>>(w_at, w_pr, enc, SS, HH);

  // 6-9) cells 1..4 (hid written directly into d_out)
  cellf(1, w_at,                 w_h0,                  hid);
  cellf(2, hid,                  hm + (size_t)BB * HH,  hid + (size_t)BB * HH);
  cellf(3, hid + (size_t)BB*HH,  hm + 2*(size_t)BB*HH,  hid + 2*(size_t)BB*HH);
  cellf(4, hid + 2*(size_t)BB*HH,hm + 3*(size_t)BB*HH,  hid + 3*(size_t)BB*HH);

  // 10) readout
  gemm(emb,                   We2o, be2o, w_r1, HH, HH); ln(w_r1, w_r1, eg, eb, BB, HH);
  gemm(hid + 3*(size_t)BB*HH, Wh2o, bh2o, w_r2, HH, HH); ln(w_r2, w_r2, hg, hb, BB, HH);
  gemm(w_at,                  Wc2o, bc2o, w_r3, HH, HH); ln(w_r3, w_r3, cg, cb, BB, HH);
  readout_kernel<<<dim3((BB * HH) / 256), dim3(256), 0, stream>>>(outp, w_r1, w_r2, w_r3);
}
